// DepthSeparableConv2d_conv2_6_37572373905417
// MI455X (gfx1250) — matrix operand order repaired and checked by reading, not compiled, not run on hardware
//
#include <hip/hip_runtime.h>

#define EPS 1e-5f
#define DW_THR 4.0f
#define PW_THR 1e-3f

typedef __attribute__((ext_vector_type(16))) __bf16 v16bf;
typedef __attribute__((ext_vector_type(8)))  float  v8f;
typedef __attribute__((ext_vector_type(4)))  unsigned int u32x4;

union Frag16 {            // 32 bytes: one 16x32 bf16 A/B fragment per lane
    u32x4 q[2];
    v16bf v;
};

static __device__ __forceinline__ unsigned short f2bf(float f) {
    unsigned u = __float_as_uint(f);
    u += 0x7FFFu + ((u >> 16) & 1u);      // round-to-nearest-even
    return (unsigned short)(u >> 16);
}

static __device__ __forceinline__ void wait_asynccnt0() {
#if defined(__has_builtin)
#if __has_builtin(__builtin_amdgcn_s_wait_asynccnt)
    __builtin_amdgcn_s_wait_asynccnt(0);
    return;
#else
    asm volatile("s_wait_asynccnt 0" ::: "memory");
    return;
#endif
#else
    asm volatile("s_wait_asynccnt 0" ::: "memory");
#endif
}

// one 16-byte async global->LDS copy (CDNA5 TDM-adjacent async path, ASYNCcnt)
static __device__ __forceinline__ void async_copy_b128(unsigned lds_byte_off,
                                                       const void* gaddr) {
    asm volatile("global_load_async_to_lds_b128 %0, %1, off"
                 :: "v"(lds_byte_off), "v"(gaddr)
                 : "memory");
}

// ---------------------------------------------------------------------------
// K0: fold BN2 scale into pointwise weights (fp32 -> bf16), build fused bias,
//     clear the per-(b,cout) running-max scratch.
// grid = 512 blocks (one per cout), 256 threads (one per cin)
// ---------------------------------------------------------------------------
__global__ void prep_kernel(const float* __restrict__ pw_w,
                            const float* __restrict__ pw_b,
                            const float* __restrict__ g2,
                            const float* __restrict__ be2,
                            const float* __restrict__ m2,
                            const float* __restrict__ v2,
                            unsigned short* __restrict__ wt,
                            float* __restrict__ b2f,
                            unsigned int* __restrict__ maxws) {
    const int o = blockIdx.x;
    const int c = threadIdx.x;
    const float inv = g2[o] * rsqrtf(v2[o] + EPS);
    wt[o * 256 + c] = f2bf(pw_w[o * 256 + c] * inv);
    if (c == 0) b2f[o] = pw_b[o] * inv + be2[o] - m2[o] * inv;
    const int id = o * 256 + c;
    if (id < 32 * 512) maxws[id] = 0u;      // relu output >= 0 -> uint max works
}

// ---------------------------------------------------------------------------
// K1: depthwise 3x3 + bias + BN1 + ReLU + per-map prune(DW_THR).
//     One block per (b,c) map; map staged in LDS; output written pruned, as
//     bf16 in [b][hw][cin] layout (GEMM-B friendly).
// grid = 32*256 blocks, 256 threads
// ---------------------------------------------------------------------------
__global__ void dw_kernel(const float* __restrict__ x,
                          const float* __restrict__ dw_w,
                          const float* __restrict__ dw_b,
                          const float* __restrict__ g1,
                          const float* __restrict__ be1,
                          const float* __restrict__ m1,
                          const float* __restrict__ v1,
                          unsigned short* __restrict__ yt) {
    __shared__ float xs[3136];
    __shared__ float red[256];

    const int bc = blockIdx.x;
    const int b  = bc >> 8;
    const int c  = bc & 255;
    const int tid = threadIdx.x;

    const float* xp = x + (size_t)bc * 3136;
    for (int i = tid; i < 3136; i += 256) xs[i] = xp[i];

    const float inv = g1[c] * rsqrtf(v1[c] + EPS);
    float wgt[9];
#pragma unroll
    for (int j = 0; j < 9; ++j) wgt[j] = dw_w[c * 9 + j] * inv;
    const float bb = dw_b[c] * inv + be1[c] - m1[c] * inv;
    __syncthreads();

    float vals[13];
    float mx = 0.0f;
#pragma unroll
    for (int i = 0; i < 13; ++i) {
        const int p = tid + i * 256;
        float r = 0.0f;
        if (p < 3136) {
            const int h = p / 56;
            const int w = p - h * 56;
            float acc = 0.0f;
#pragma unroll
            for (int dy = -1; dy <= 1; ++dy) {
#pragma unroll
                for (int dx = -1; dx <= 1; ++dx) {
                    const int hh = h + dy, ww = w + dx;
                    if (hh >= 0 && hh < 56 && ww >= 0 && ww < 56)
                        acc += xs[hh * 56 + ww] * wgt[(dy + 1) * 3 + (dx + 1)];
                }
            }
            r = fmaxf(acc + bb, 0.0f);
        }
        vals[i] = r;
        mx = fmaxf(mx, r);
    }

    red[tid] = mx;
    __syncthreads();
#pragma unroll
    for (int s = 128; s > 0; s >>= 1) {
        if (tid < s) red[tid] = fmaxf(red[tid], red[tid + s]);
        __syncthreads();
    }
    const bool keep = red[0] >= DW_THR;

    unsigned short* yp = yt + (size_t)b * 3136 * 256 + c;
#pragma unroll
    for (int i = 0; i < 13; ++i) {
        const int p = tid + i * 256;
        if (p < 3136) yp[(size_t)p * 256] = keep ? f2bf(vals[i]) : (unsigned short)0;
    }
}

// ---------------------------------------------------------------------------
// K2: pointwise GEMM via v_wmma_f32_16x16x32_bf16 with double-buffered
//     async global->LDS staging (ASYNCcnt).
// grid = (28 hw-tiles, 4 cout-blocks, 32 batch), 256 threads (8 waves)
// wave tile: 16(M) x 112(N), block tile: 128(M) x 112(N), K = 256 in 8 steps
// ---------------------------------------------------------------------------
__global__ void pw_kernel(const unsigned short* __restrict__ yt,
                          const unsigned short* __restrict__ wt,
                          const float* __restrict__ b2f,
                          float* __restrict__ zout,
                          unsigned int* __restrict__ maxws) {
    // double-buffered B tile: 112 rows (hw) x 32 halves (cin), row stride 40
    __shared__ unsigned short Bs[2][112 * 40];

    const int hw0   = blockIdx.x * 112;
    const int cout0 = blockIdx.y * 128;
    const int b     = blockIdx.z;
    const int tid   = threadIdx.x;
    const int wave  = tid >> 5;
    const int lane  = tid & 31;
    const int lmod  = lane & 15;
    const int lhi   = lane >> 4;           // 0 or 1

    v8f acc[7] = {};

    // A fragment addressing (ISA 16-bit A 16x32 layout):
    //   m = lane%16 ; halves 0..7 -> k = 8*lhi + j ; halves 8..15 -> k += 8
    const int m_a = cout0 + wave * 16 + lmod;
    const unsigned short* Arow = wt + (size_t)m_a * 256;
    const int kbaseA = lhi * 8;

    // B fragment addressing (ISA 16-bit B 32x16 layout):
    //   n = lane%16 ; halves 0..7 -> k = 8*lhi + j ; halves 8..15 -> k += 8
    const int kb2 = lhi * 8;

    const unsigned short* ytb = yt + ((size_t)b * 3136 + hw0) * 256;

    // LDS byte offsets of the two buffers (generic LDS ptr: low 32 bits = offset)
    const unsigned base0 = (unsigned)(size_t)(&Bs[0][0]);
    const unsigned base1 = (unsigned)(size_t)(&Bs[1][0]);

    // stage one K-slice (112 rows x 64 B = 448 x 16-byte chunks) into a buffer
    auto issue_slice = [&](unsigned ldsbase, int k0) {
        {
            const int row = tid >> 2, quad = tid & 3;
            async_copy_b128(ldsbase + (unsigned)(row * 80 + quad * 16),
                            ytb + (size_t)row * 256 + k0 + quad * 8);
        }
        if (tid < 192) {
            const int ch = tid + 256;
            const int row = ch >> 2, quad = ch & 3;
            async_copy_b128(ldsbase + (unsigned)(row * 80 + quad * 16),
                            ytb + (size_t)row * 256 + k0 + quad * 8);
        }
    };

    // prologue: fill buffer 0 with K-step 0
    issue_slice(base0, 0);
    wait_asynccnt0();
    __syncthreads();

    for (int ks = 0; ks < 8; ++ks) {
        const int k0 = ks * 32;
        // kick off next K-slice into the other buffer while we compute
        if (ks < 7) issue_slice((ks & 1) ? base0 : base1, k0 + 32);

        const unsigned short* Bcur = &Bs[ks & 1][0];

        Frag16 a;
        a.q[0] = *(const u32x4*)(Arow + k0 + kbaseA);
        a.q[1] = *(const u32x4*)(Arow + k0 + kbaseA + 16);

#pragma unroll
        for (int nt = 0; nt < 7; ++nt) {
            Frag16 bf;
            const u32x4* bp = (const u32x4*)(Bcur + (nt * 16 + lmod) * 40 + kb2);
            bf.q[0] = bp[0];
            bf.q[1] = bp[2];
            acc[nt] = __builtin_amdgcn_wmma_f32_16x16x32_bf16(
                false, a.v, false, bf.v, (short)0, acc[nt], false, false);
        }

        if (ks < 7) {
            wait_asynccnt0();     // own async copies landed in LDS
            __syncthreads();      // everyone's copies landed, readers done
        }
    }

    // epilogue: C/D layout  m = rowbase + r , n = lane%16
    const int rowbase = cout0 + wave * 16 + lhi * 8;
    float bias[8], mxr[8];
#pragma unroll
    for (int r = 0; r < 8; ++r) { bias[r] = b2f[rowbase + r]; mxr[r] = 0.0f; }

    float* zb = zout + (size_t)b * 512 * 3136;
#pragma unroll
    for (int nt = 0; nt < 7; ++nt) {
        const int col = hw0 + nt * 16 + lmod;
#pragma unroll
        for (int r = 0; r < 8; ++r) {
            const float vv = fmaxf(acc[nt][r] + bias[r], 0.0f);
            zb[(size_t)(rowbase + r) * 3136 + col] = vv;
            mxr[r] = fmaxf(mxr[r], vv);
        }
    }

    // max over the 16 lanes of this half-group (xor masks < 16 stay in group)
#pragma unroll
    for (int off = 8; off > 0; off >>= 1) {
#pragma unroll
        for (int r = 0; r < 8; ++r)
            mxr[r] = fmaxf(mxr[r], __shfl_xor(mxr[r], off, 32));
    }
    if (lmod == 0) {
#pragma unroll
        for (int r = 0; r < 8; ++r)
            atomicMax(&maxws[b * 512 + rowbase + r], __float_as_uint(mxr[r]));
    }
}

// ---------------------------------------------------------------------------
// K3: prune2 — zero any (b,cout) map whose relu-max < PW_THR.
// grid = 32*512 blocks, 256 threads
// ---------------------------------------------------------------------------
__global__ void prune2_kernel(const unsigned int* __restrict__ maxws,
                              float* __restrict__ zout) {
    const int bc = blockIdx.x;
    if (__uint_as_float(maxws[bc]) >= PW_THR) return;
    float* zp = zout + (size_t)bc * 3136;
    for (int i = threadIdx.x; i < 3136; i += 256) zp[i] = 0.0f;
}

// ---------------------------------------------------------------------------
extern "C" void kernel_launch(void* const* d_in, const int* in_sizes, int n_in,
                              void* d_out, int out_size, void* d_ws, size_t ws_size,
                              hipStream_t stream) {
    const float* x    = (const float*)d_in[0];
    const float* dw_w = (const float*)d_in[1];
    const float* dw_b = (const float*)d_in[2];
    const float* g1   = (const float*)d_in[3];
    const float* be1  = (const float*)d_in[4];
    const float* m1   = (const float*)d_in[5];
    const float* v1   = (const float*)d_in[6];
    const float* pw_w = (const float*)d_in[7];
    const float* pw_b = (const float*)d_in[8];
    const float* g2   = (const float*)d_in[9];
    const float* be2  = (const float*)d_in[10];
    const float* m2   = (const float*)d_in[11];
    const float* v2   = (const float*)d_in[12];
    float* z = (float*)d_out;

    char* ws = (char*)d_ws;
    const size_t YT_BYTES = (size_t)32 * 3136 * 256 * 2;  // 51,380,224
    const size_t WT_BYTES = (size_t)512 * 256 * 2;        //    262,144
    const size_t BF_BYTES = (size_t)512 * 4;              //      2,048
    unsigned short* yt    = (unsigned short*)ws;
    unsigned short* wt    = (unsigned short*)(ws + YT_BYTES);
    float*          b2f   = (float*)(ws + YT_BYTES + WT_BYTES);
    unsigned int*   maxws = (unsigned int*)(ws + YT_BYTES + WT_BYTES + BF_BYTES);

    prep_kernel<<<512, 256, 0, stream>>>(pw_w, pw_b, g2, be2, m2, v2, wt, b2f, maxws);
    dw_kernel<<<32 * 256, 256, 0, stream>>>(x, dw_w, dw_b, g1, be1, m1, v1, yt);
    pw_kernel<<<dim3(28, 4, 32), 256, 0, stream>>>(yt, wt, b2f, z, maxws);
    prune2_kernel<<<32 * 512, 256, 0, stream>>>(maxws, z);
}
